// MixtureOfDepths_75041668596054
// MI455X (gfx1250) — compile-verified, matrix operand order checked by reading, not run on hardware
//
#include <hip/hip_runtime.h>
#include <hip/hip_bf16.h>

// ---------------- problem constants ----------------
#define DIM   2048
#define HID   8192
#define NTOK  8192
#define KSEL  4096   // int(NTOK * 0.5)
#define EPSV  1e-6f

typedef __attribute__((ext_vector_type(16))) __bf16 v16bf;
typedef __attribute__((ext_vector_type(8)))  float  v8f;

typedef int v4i __attribute__((vector_size(16)));
typedef __attribute__((address_space(1))) void gas_void;   // global
typedef __attribute__((address_space(3))) void las_void;   // LDS
typedef __attribute__((address_space(1))) v4i  gas_v4i;
typedef __attribute__((address_space(3))) v4i  las_v4i;

__device__ __forceinline__ unsigned short f2bf(float f) {
    unsigned u = __float_as_uint(f);
    unsigned r = (u + 0x7FFFu + ((u >> 16) & 1u)) >> 16;   // round-to-nearest-even
    return (unsigned short)r;
}

// ---------------- CDNA5 async global->LDS (ASYNCcnt-tracked) ----------------
__device__ __forceinline__ void async_cp16(void* lds_dst, const void* gsrc) {
#if __has_builtin(__builtin_amdgcn_global_load_async_to_lds_b128)
    __builtin_amdgcn_global_load_async_to_lds_b128(
        (gas_v4i*)(gas_void*)gsrc, (las_v4i*)(las_void*)lds_dst, 0, 0);
#else
    asm volatile("global_load_async_to_lds_b128 %0, %1, off"
                 :: "v"((unsigned)(unsigned long long)(las_void*)lds_dst),
                    "v"(gsrc)
                 : "memory");
#endif
}

__device__ __forceinline__ void async_wait0() {
#if __has_builtin(__builtin_amdgcn_s_wait_asynccnt)
    __builtin_amdgcn_s_wait_asynccnt(0);
#else
    asm volatile("s_wait_asynccnt 0x0" ::: "memory");
#endif
}

// =====================================================================
// Kernel 1: fused RMSNorm + router logit. One block (256 thr) per token.
// =====================================================================
__global__ __launch_bounds__(256)
void k_rmsnorm_router(const float* __restrict__ x,
                      const float* __restrict__ nw,
                      const float* __restrict__ rw,
                      const float* __restrict__ rb,
                      unsigned short* __restrict__ xn,
                      float* __restrict__ logits)
{
    const int row = blockIdx.x;
    const int t   = threadIdx.x;
    const float* xr = x + (size_t)row * DIM;

    float v[8];
    float ss = 0.f;
#pragma unroll
    for (int i = 0; i < 8; ++i) { v[i] = xr[t + i * 256]; ss += v[i] * v[i]; }

    __shared__ float red[256];
    red[t] = ss; __syncthreads();
    for (int s = 128; s > 0; s >>= 1) { if (t < s) red[t] += red[t + s]; __syncthreads(); }
    const float rms = rsqrtf(red[0] / (float)DIM + EPSV);
    __syncthreads();

    float dot = 0.f;
#pragma unroll
    for (int i = 0; i < 8; ++i) {
        const int c = t + i * 256;
        const float xv = v[i] * rms * nw[c];
        xn[(size_t)row * DIM + c] = f2bf(xv);
        dot += xv * rw[c];
    }
    red[t] = dot; __syncthreads();
    for (int s = 128; s > 0; s >>= 1) { if (t < s) red[t] += red[t + s]; __syncthreads(); }
    if (t == 0) logits[row] = red[0] + rb[0];
}

// =====================================================================
// Kernel 2: exact k-of-8192 select, single block, MSB-first radix select.
// =====================================================================
__global__ __launch_bounds__(1024)
void k_topk_select(const float* __restrict__ logits, int* __restrict__ sel)
{
    __shared__ unsigned keys[NTOK];
    __shared__ unsigned hist[16];
    __shared__ unsigned sPrefix, sR, sCnt, sTie;
    const int t = threadIdx.x;

    for (int i = t; i < NTOK; i += 1024) {
        unsigned f = __float_as_uint(logits[i]);
        keys[i] = (f & 0x80000000u) ? ~f : (f | 0x80000000u);
    }
    if (t == 0) { sPrefix = 0u; sR = (unsigned)KSEL; sCnt = 0u; sTie = 0u; }
    __syncthreads();

    for (int shift = 28; shift >= 0; shift -= 4) {
        if (t < 16) hist[t] = 0u;
        __syncthreads();
        const unsigned mask = (shift == 28) ? 0u : (0xFFFFFFFFu << (shift + 4));
        const unsigned pref = sPrefix;
        for (int i = t; i < NTOK; i += 1024) {
            unsigned u = keys[i];
            if ((u & mask) == pref) atomicAdd(&hist[(u >> shift) & 15u], 1u);
        }
        __syncthreads();
        if (t == 0) {
            unsigned r = sR;
            int d = 15;
            while (d > 0 && r > hist[d]) { r -= hist[d]; --d; }
            sPrefix = pref | ((unsigned)d << shift);
            sR = r;
        }
        __syncthreads();
    }
    const unsigned T = sPrefix, rt = sR;
    for (int i = t; i < NTOK; i += 1024) {
        const unsigned u = keys[i];
        if (u > T) {
            unsigned p = atomicAdd(&sCnt, 1u);
            sel[p] = i;
        } else if (u == T) {
            unsigned tk = atomicAdd(&sTie, 1u);
            if (tk < rt) { unsigned p = atomicAdd(&sCnt, 1u); sel[p] = i; }
        }
    }
}

// =====================================================================
// Kernel 3: transpose + fp32->bf16. src[R][C] -> dst[C][R] (weights N-major).
// =====================================================================
__global__ __launch_bounds__(256)
void k_transpose_bf16(const float* __restrict__ src, unsigned short* __restrict__ dst,
                      int R, int C)
{
    __shared__ float tile[32][33];
    const int c0 = blockIdx.x * 32, r0 = blockIdx.y * 32;
    const int tx = threadIdx.x, ty = threadIdx.y;     // 32 x 8
#pragma unroll
    for (int j = 0; j < 32; j += 8)
        tile[ty + j][tx] = src[(size_t)(r0 + ty + j) * C + (c0 + tx)];
    __syncthreads();
#pragma unroll
    for (int j = 0; j < 32; j += 8)
        dst[(size_t)(c0 + ty + j) * R + (r0 + tx)] = f2bf(tile[tx][ty + j]);
}

// =====================================================================
// Kernel 4: out = x
// =====================================================================
__global__ void k_copy_out(const float4* __restrict__ src, float4* __restrict__ dst, int n)
{
    int i = blockIdx.x * blockDim.x + threadIdx.x;
    const int stride = gridDim.x * blockDim.x;
    for (; i < n; i += stride) dst[i] = src[i];
}

// ---------------- WMMA fragment helper ----------------
// A (16x32 bf16) per-lane: row = lane%16, K chunks {8h..8h+7} U {16+8h..}, h=lane>>4.
// B staged N-major in LDS -> identical addressing.
__device__ __forceinline__ v16bf load_frag(const unsigned short* p, int kh)
{
    union { v16bf v; uint4 q[2]; } u;
    u.q[0] = *(const uint4*)(p + kh);
    u.q[1] = *(const uint4*)(p + 16 + kh);
    return u.v;
}

#define LDSPAD 40  // 32 + 8 bf16 pad: 80B row stride, conflict-free 16B lane loads

// =====================================================================
// Kernel 5: GEMM1 + GELU.  h = gelu(xn[sel] @ w1t^T + b1), bf16 out.
// Block tile 128Mx256N, K-step 32. 8 waves (2Mx4N), each 64x64 via 4x4
// v_wmma_f32_16x16x32_bf16. Async global->LDS double buffering,
// one barrier per K-step.
// =====================================================================
__global__ __launch_bounds__(256)
void k_gemm1_gelu(const unsigned short* __restrict__ xn,   // [NTOK, DIM] bf16
                  const unsigned short* __restrict__ w1t,  // [HID, DIM]  bf16
                  const float* __restrict__ b1,
                  const int* __restrict__ sel,
                  unsigned short* __restrict__ hout)       // [KSEL, HID] bf16
{
    const int n0 = blockIdx.x * 256;
    const int m0 = blockIdx.y * 128;

    __shared__ __align__(16) unsigned short As[2][128][LDSPAD];
    __shared__ __align__(16) unsigned short Bs[2][256][LDSPAD];
    __shared__ int rows[128];

    const int t = threadIdx.x;
    if (t < 128) rows[t] = sel[m0 + t];
    __syncthreads();

    const int ar = t >> 1, aseg = (t & 1) * 16;    // A staging: 2 thr/row, 32B each
    const size_t arow = (size_t)rows[ar] * DIM;
    const size_t brow = (size_t)(n0 + t) * DIM;    // B staging: 1 thr/row, 64B

    const int lane = t & 31, wave = t >> 5;
    const int wm = (wave >> 2) * 64;               // wave M origin (0/64)
    const int wn = (wave & 3) * 64;                // wave N origin (0/64/128/192)
    const int fr = lane & 15;
    const int kh = (lane >> 4) * 8;

    v8f acc[4][4];
    const v8f vz = {0.f,0.f,0.f,0.f,0.f,0.f,0.f,0.f};
#pragma unroll
    for (int i = 0; i < 4; ++i)
#pragma unroll
        for (int j = 0; j < 4; ++j) acc[i][j] = vz;

    // prologue: async tile 0 -> buffer 0
    {
        const unsigned short* ga = xn + arow + aseg;
        async_cp16(&As[0][ar][aseg],     ga);
        async_cp16(&As[0][ar][aseg + 8], ga + 8);
        const unsigned short* gb = w1t + brow;
        async_cp16(&Bs[0][t][0],  gb);
        async_cp16(&Bs[0][t][8],  gb + 8);
        async_cp16(&Bs[0][t][16], gb + 16);
        async_cp16(&Bs[0][t][24], gb + 24);
    }

    int buf = 0;
    for (int k0 = 0; k0 < DIM; k0 += 32) {
        async_wait0();        // my share of current tile landed in LDS
        __syncthreads();      // everyone's share landed; prev buffer free

        const int k1 = k0 + 32;
        if (k1 < DIM) {
            const int nb = buf ^ 1;
            const unsigned short* ga = xn + arow + k1 + aseg;
            async_cp16(&As[nb][ar][aseg],     ga);
            async_cp16(&As[nb][ar][aseg + 8], ga + 8);
            const unsigned short* gb = w1t + brow + k1;
            async_cp16(&Bs[nb][t][0],  gb);
            async_cp16(&Bs[nb][t][8],  gb + 8);
            async_cp16(&Bs[nb][t][16], gb + 16);
            async_cp16(&Bs[nb][t][24], gb + 24);
        }

        v16bf a[4], b[4];
#pragma unroll
        for (int i = 0; i < 4; ++i) a[i] = load_frag(&As[buf][wm + i * 16 + fr][0], kh);
#pragma unroll
        for (int j = 0; j < 4; ++j) b[j] = load_frag(&Bs[buf][wn + j * 16 + fr][0], kh);
#pragma unroll
        for (int i = 0; i < 4; ++i)
#pragma unroll
            for (int j = 0; j < 4; ++j)
                acc[i][j] = __builtin_amdgcn_wmma_f32_16x16x32_bf16(
                    false, a[i], false, b[j], (short)0, acc[i][j], false, false);
        buf ^= 1;
    }

    // epilogue: exact GELU, store bf16
#pragma unroll
    for (int i = 0; i < 4; ++i) {
        const int gm = m0 + wm + i * 16 + kh;
#pragma unroll
        for (int j = 0; j < 4; ++j) {
            const int gn = n0 + wn + j * 16 + fr;
            const float bias = b1[gn];
#pragma unroll
            for (int e = 0; e < 8; ++e) {
                float v = acc[i][j][e] + bias;
                v = 0.5f * v * (1.0f + erff(v * 0.70710678118654752f));
                hout[(size_t)(gm + e) * HID + gn] = f2bf(v);
            }
        }
    }
}

// =====================================================================
// Kernel 6: GEMM2 + fused residual/gamma scatter.
// =====================================================================
__global__ __launch_bounds__(256)
void k_gemm2_scatter(const unsigned short* __restrict__ hbuf, // [KSEL, HID] bf16
                     const unsigned short* __restrict__ w2t,  // [DIM, HID]  bf16
                     const float* __restrict__ b2,
                     const float* __restrict__ gamma,
                     const int* __restrict__ sel,
                     const float* __restrict__ x,
                     float* __restrict__ out)
{
    const int n0 = blockIdx.x * 256;
    const int m0 = blockIdx.y * 128;

    __shared__ __align__(16) unsigned short As[2][128][LDSPAD];
    __shared__ __align__(16) unsigned short Bs[2][256][LDSPAD];
    __shared__ int rows[128];

    const int t = threadIdx.x;
    if (t < 128) rows[t] = sel[m0 + t];
    __syncthreads();

    const int ar = t >> 1, aseg = (t & 1) * 16;
    const size_t arow = (size_t)(m0 + ar) * HID;   // A compacted: no gather
    const size_t brow = (size_t)(n0 + t) * HID;

    const int lane = t & 31, wave = t >> 5;
    const int wm = (wave >> 2) * 64;
    const int wn = (wave & 3) * 64;
    const int fr = lane & 15;
    const int kh = (lane >> 4) * 8;

    v8f acc[4][4];
    const v8f vz = {0.f,0.f,0.f,0.f,0.f,0.f,0.f,0.f};
#pragma unroll
    for (int i = 0; i < 4; ++i)
#pragma unroll
        for (int j = 0; j < 4; ++j) acc[i][j] = vz;

    {
        const unsigned short* ga = hbuf + arow + aseg;
        async_cp16(&As[0][ar][aseg],     ga);
        async_cp16(&As[0][ar][aseg + 8], ga + 8);
        const unsigned short* gb = w2t + brow;
        async_cp16(&Bs[0][t][0],  gb);
        async_cp16(&Bs[0][t][8],  gb + 8);
        async_cp16(&Bs[0][t][16], gb + 16);
        async_cp16(&Bs[0][t][24], gb + 24);
    }

    int buf = 0;
    for (int k0 = 0; k0 < HID; k0 += 32) {
        async_wait0();
        __syncthreads();

        const int k1 = k0 + 32;
        if (k1 < HID) {
            const int nb = buf ^ 1;
            const unsigned short* ga = hbuf + arow + k1 + aseg;
            async_cp16(&As[nb][ar][aseg],     ga);
            async_cp16(&As[nb][ar][aseg + 8], ga + 8);
            const unsigned short* gb = w2t + brow + k1;
            async_cp16(&Bs[nb][t][0],  gb);
            async_cp16(&Bs[nb][t][8],  gb + 8);
            async_cp16(&Bs[nb][t][16], gb + 16);
            async_cp16(&Bs[nb][t][24], gb + 24);
        }

        v16bf a[4], b[4];
#pragma unroll
        for (int i = 0; i < 4; ++i) a[i] = load_frag(&As[buf][wm + i * 16 + fr][0], kh);
#pragma unroll
        for (int j = 0; j < 4; ++j) b[j] = load_frag(&Bs[buf][wn + j * 16 + fr][0], kh);
#pragma unroll
        for (int i = 0; i < 4; ++i)
#pragma unroll
            for (int j = 0; j < 4; ++j)
                acc[i][j] = __builtin_amdgcn_wmma_f32_16x16x32_bf16(
                    false, a[i], false, b[j], (short)0, acc[i][j], false, false);
        buf ^= 1;
    }

    // epilogue: out[tok] = x[tok] + (acc + b2) * gamma, scatter by token id
#pragma unroll
    for (int i = 0; i < 4; ++i) {
        const int lm = wm + i * 16 + kh;
#pragma unroll
        for (int j = 0; j < 4; ++j) {
            const int gn = n0 + wn + j * 16 + fr;
            const float bias = b2[gn];
            const float g    = gamma[gn];
#pragma unroll
            for (int e = 0; e < 8; ++e) {
                const int tok = rows[lm + e];
                const size_t off = (size_t)tok * DIM + gn;
                out[off] = x[off] + (acc[i][j][e] + bias) * g;
            }
        }
    }
}

// =====================================================================
// host launcher
// =====================================================================
extern "C" void kernel_launch(void* const* d_in, const int* in_sizes, int n_in,
                              void* d_out, int out_size, void* d_ws, size_t ws_size,
                              hipStream_t stream)
{
    (void)in_sizes; (void)n_in; (void)out_size; (void)ws_size;
    const float* x     = (const float*)d_in[0];
    const float* nw    = (const float*)d_in[1];
    const float* rw    = (const float*)d_in[2];
    const float* rb    = (const float*)d_in[3];
    const float* w1    = (const float*)d_in[4];
    const float* b1    = (const float*)d_in[5];
    const float* w2    = (const float*)d_in[6];
    const float* b2    = (const float*)d_in[7];
    const float* gamma = (const float*)d_in[8];
    float* out = (float*)d_out;

    char* ws = (char*)d_ws;
    unsigned short* xn   = (unsigned short*)(ws);                       // 32 MB
    unsigned short* w1t  = (unsigned short*)(ws + (size_t)33554432);    // 32 MB
    unsigned short* w2t  = (unsigned short*)(ws + (size_t)67108864);    // 32 MB
    unsigned short* hbuf = (unsigned short*)(ws + (size_t)100663296);   // 64 MB
    float*          lg   = (float*)(ws + (size_t)167772160);            // 32 KB
    int*            sel  = (int*)(ws + (size_t)167804928);              // 16 KB

    k_rmsnorm_router<<<NTOK, 256, 0, stream>>>(x, nw, rw, rb, xn, lg);
    k_transpose_bf16<<<dim3(HID / 32, DIM / 32), dim3(32, 8), 0, stream>>>(w1, w1t, DIM, HID);
    k_transpose_bf16<<<dim3(DIM / 32, HID / 32), dim3(32, 8), 0, stream>>>(w2, w2t, HID, DIM);
    k_topk_select<<<1, 1024, 0, stream>>>(lg, sel);
    k_copy_out<<<4096, 256, 0, stream>>>((const float4*)x, (float4*)out, NTOK * DIM / 4);
    k_gemm1_gelu<<<dim3(HID / 256, KSEL / 128), 256, 0, stream>>>(xn, w1t, b1, sel, hbuf);
    k_gemm2_scatter<<<dim3(DIM / 256, KSEL / 128), 256, 0, stream>>>(hbuf, w2t, b2, gamma, sel, x, out);
}
